// AttentionHead_11613591568736
// MI455X (gfx1250) — matrix operand order repaired and checked by reading, not compiled, not run on hardware
//
#include <hip/hip_runtime.h>

typedef __attribute__((ext_vector_type(16))) __bf16 v16bf;
typedef __attribute__((ext_vector_type(8)))  __bf16 v8bf;
typedef __attribute__((ext_vector_type(8)))  float  v8f;
typedef __attribute__((ext_vector_type(4)))  float  v4f;

#define B_DIM 4
#define T_DIM 4096
#define D_DIM 1024
#define H_DIM 64
#define NEG_INF (-1.0e9f)
#define SCALE (0.03125f)   /* 1/sqrt(1024), exact power of two */

static __device__ __forceinline__ v16bf cat8(v8bf a, v8bf b) {
  return __builtin_shufflevector(a, b, 0,1,2,3,4,5,6,7,8,9,10,11,12,13,14,15);
}
static __device__ __forceinline__ v8f wmma_bf16(v16bf a, v16bf b, v8f c) {
  return __builtin_amdgcn_wmma_f32_16x16x32_bf16(false, a, false, b, (short)0, c, false, false);
}

// ---------------- Kernel 1: convert weights f32 -> bf16 ----------------
__global__ void __launch_bounds__(256)
wconv_kernel(const float* __restrict__ Wq, const float* __restrict__ Wk,
             const float* __restrict__ Wv,
             __bf16* __restrict__ oq, __bf16* __restrict__ ok,
             __bf16* __restrict__ ov) {
  int i = blockIdx.x * 256 + threadIdx.x;
  if (i < H_DIM * D_DIM) {
    oq[i] = (__bf16)Wq[i];
    ok[i] = (__bf16)Wk[i];
    ov[i] = (__bf16)Wv[i];
  }
}

// ---------------- Kernel 1b: pack mask int32 -> bit matrix [T, T/32] ----------------
__global__ void __launch_bounds__(256)
mpack_kernel(const int* __restrict__ mask, unsigned* __restrict__ mbits) {
  int w = blockIdx.x * 256 + threadIdx.x;          // word index
  if (w < T_DIM * T_DIM / 32) {
    const int* mp = mask + (size_t)w * 32;
    unsigned bits = 0;
#pragma unroll
    for (int i = 0; i < 32; ++i) bits |= (mp[i] != 0 ? 1u : 0u) << i;
    mbits[w] = bits;
  }
}

// ---------------- Kernel 2: fused QKV projection (WMMA bf16, double pipeline) ----------------
// One wave = 16 rows of x -> 16x64 of Q, K, V.
// x (HBM stream) pipelined one d0-step ahead; weight B-frags pipelined one
// nt-step ahead. sched_group_barrier enforces the pipelined order pre-RA:
//   VMEM_READx10, (WMMAx3, VMEM_READx6)x3, WMMAx3
__global__ void __launch_bounds__(128)
qkv_proj_kernel(const float* __restrict__ x,
                const __bf16* __restrict__ Wqb, const __bf16* __restrict__ Wkb,
                const __bf16* __restrict__ Wvb,
                __bf16* __restrict__ Qo, __bf16* __restrict__ Ko,
                __bf16* __restrict__ Vto) {
  const int lane = threadIdx.x & 31;
  const int wid  = threadIdx.x >> 5;
  const int tile = blockIdx.x * 4 + wid;          // 0..1023
  const int row0 = tile * 16;                     // flat row over B*T
  const int nn   = lane & 15;
  const int g    = lane >> 4;

  const v8f vz = {0.f,0.f,0.f,0.f,0.f,0.f,0.f,0.f};
  v8f aq[4], ak[4], av[4];
#pragma unroll
  for (int nt = 0; nt < 4; ++nt) { aq[nt] = vz; ak[nt] = vz; av[nt] = vz; }

  const float* xr = x + (size_t)(row0 + nn) * D_DIM;

  // x pipeline: first A-frag preloaded
  v4f f0 = *(const v4f*)(xr + 16*g);
  v4f f1 = *(const v4f*)(xr + 16*g + 4);
  v4f f2 = *(const v4f*)(xr + 16*g + 8);
  v4f f3 = *(const v4f*)(xr + 16*g + 12);

  for (int d0 = 0; d0 < D_DIM; d0 += 32) {
    // prefetch next iteration's x (HBM latency hides under 12 WMMAs)
    const int dn = (d0 + 32 < D_DIM) ? (d0 + 32) : d0;   // clamped prefetch
    const float* xn = xr + dn;
    v4f n0 = *(const v4f*)(xn + 16*g);
    v4f n1 = *(const v4f*)(xn + 16*g + 4);
    v4f n2 = *(const v4f*)(xn + 16*g + 8);
    v4f n3 = *(const v4f*)(xn + 16*g + 12);

    const int co = d0 + 16*g;                     // B-frag: lane=N, elems -> K=16g+j
    const size_t wrow0 = (size_t)nn * D_DIM + co;

    // prologue: nt=0 B-frags
    v16bf cq = *(const v16bf*)(Wqb + wrow0);
    v16bf ck = *(const v16bf*)(Wkb + wrow0);
    v16bf cv = *(const v16bf*)(Wvb + wrow0);

    // A-frag (16x32 bf16): lane=M, elems 0..15 -> K=16g+j
    v16bf af;
    af[0]=(__bf16)f0[0];  af[1]=(__bf16)f0[1];  af[2]=(__bf16)f0[2];  af[3]=(__bf16)f0[3];
    af[4]=(__bf16)f1[0];  af[5]=(__bf16)f1[1];  af[6]=(__bf16)f1[2];  af[7]=(__bf16)f1[3];
    af[8]=(__bf16)f2[0];  af[9]=(__bf16)f2[1];  af[10]=(__bf16)f2[2]; af[11]=(__bf16)f2[3];
    af[12]=(__bf16)f3[0]; af[13]=(__bf16)f3[1]; af[14]=(__bf16)f3[2]; af[15]=(__bf16)f3[3];

    // pipelined nt loop: load nt+1's fragments, then run nt's 3 WMMAs
#pragma unroll
    for (int nt = 0; nt < 4; ++nt) {
      v16bf nq, nk, nv;
      if (nt < 3) {
        const size_t wrow = (size_t)((nt + 1) * 16 + nn) * D_DIM + co;
        nq = *(const v16bf*)(Wqb + wrow);
        nk = *(const v16bf*)(Wkb + wrow);
        nv = *(const v16bf*)(Wvb + wrow);
      }
      aq[nt] = wmma_bf16(af, cq, aq[nt]);
      ak[nt] = wmma_bf16(af, ck, ak[nt]);
      av[nt] = wmma_bf16(af, cv, av[nt]);
      if (nt < 3) { cq = nq; ck = nk; cv = nv; }
    }

    // enforce pipelined schedule: loads for frag-set i+1 issue before WMMAs of set i
    __builtin_amdgcn_sched_group_barrier(0x020, 10, 0);  // VMEM read x10 (x4 + nt0 frags)
    __builtin_amdgcn_sched_group_barrier(0x008, 3, 0);   // WMMA x3 (nt0)
    __builtin_amdgcn_sched_group_barrier(0x020, 6, 0);   // VMEM read x6 (nt1 frags)
    __builtin_amdgcn_sched_group_barrier(0x008, 3, 0);   // WMMA x3 (nt1)
    __builtin_amdgcn_sched_group_barrier(0x020, 6, 0);   // VMEM read x6 (nt2 frags)
    __builtin_amdgcn_sched_group_barrier(0x008, 3, 0);   // WMMA x3 (nt2)
    __builtin_amdgcn_sched_group_barrier(0x020, 6, 0);   // VMEM read x6 (nt3 frags)
    __builtin_amdgcn_sched_group_barrier(0x008, 3, 0);   // WMMA x3 (nt3)

    f0 = n0; f1 = n1; f2 = n2; f3 = n3;
  }

  const int b    = row0 >> 12;       // row0 / 4096
  const int tloc = row0 & 4095;
#pragma unroll
  for (int nt = 0; nt < 4; ++nt) {
#pragma unroll
    for (int r = 0; r < 8; ++r) {
      const int rr = r + 8*g;        // C layout: VGPR r -> M = r + 8*halfwave
      const size_t orow = (size_t)(row0 + rr) * H_DIM + nt*16 + nn;
      Qo[orow] = (__bf16)(aq[nt][r] * SCALE);   // fold softmax scale into Q
      Ko[orow] = (__bf16)ak[nt][r];
      Vto[((size_t)b * H_DIM + nt*16 + nn) * T_DIM + tloc + rr] = (__bf16)av[nt][r];
    }
  }
}

// ---------------- Kernel 3: flash attention (online softmax, WMMA) ----------------
// One wave = 16 query rows; 64 keys per iteration (amortizes softmax reductions,
// alpha-rescale and LDS round trip over 2x the keys vs a 32-key step).
__global__ void __launch_bounds__(128)
flash_attn_kernel(const __bf16* __restrict__ Q, const __bf16* __restrict__ K,
                  const __bf16* __restrict__ Vt, const unsigned* __restrict__ mbits,
                  float* __restrict__ out) {
  __shared__ __bf16 pbuf[4][16][64];   // per-wave P tile (bf16), 8 KB

  const int lane = threadIdx.x & 31;
  const int wid  = threadIdx.x >> 5;
  const int tile = blockIdx.x * 4 + wid;          // 0..1023
  const int b    = tile >> 8;                     // 256 tiles per batch
  const int t0   = (tile & 255) << 4;
  const int nn   = lane & 15;
  const int g    = lane >> 4;

  // Q A-frags (rows t0..t0+15, K-dim = h: qa0 -> h 0..31, qa1 -> h 32..63)
  const __bf16* qr = Q + ((size_t)(b * T_DIM + t0 + nn)) * H_DIM;
  v16bf qa0 = cat8(*(const v8bf*)(qr + 16*g),      *(const v8bf*)(qr + 8 + 16*g));
  v16bf qa1 = cat8(*(const v8bf*)(qr + 32 + 16*g), *(const v8bf*)(qr + 40 + 16*g));

  const v8f vz = {0.f,0.f,0.f,0.f,0.f,0.f,0.f,0.f};
  v8f o[4];
#pragma unroll
  for (int nt = 0; nt < 4; ++nt) o[nt] = vz;
  float mrow[8], lrow[8];
#pragma unroll
  for (int r = 0; r < 8; ++r) { mrow[r] = -3.0e38f; lrow[r] = 0.f; }

  const __bf16* kb = K  + (size_t)b * T_DIM * H_DIM;
  const __bf16* vb = Vt + (size_t)b * H_DIM * T_DIM;
  const unsigned long long* mrow0 =
      (const unsigned long long*)(mbits + (size_t)(t0 + 8*g) * (T_DIM / 32));

  for (int s0 = 0; s0 < T_DIM; s0 += 64) {
    // ---- mask words: one u64 per row covers all 64 keys of this step ----
    unsigned long long mw[8];
    const unsigned long long* mbp = mrow0 + (s0 >> 6);
#pragma unroll
    for (int r = 0; r < 8; ++r) mw[r] = mbp[(size_t)r * (T_DIM / 64)];

    // V B-frags, first half (keys s0..s0+31), hides under softmax VALU
    v16bf vfA[4];
#pragma unroll
    for (int nt = 0; nt < 4; ++nt)
      vfA[nt] = *(const v16bf*)(vb + (size_t)(nt*16 + nn) * T_DIM + s0 + 16*g);

    // ---- S tiles 0,1 (keys s0..s0+31) ----
    const __bf16* kp0 = kb + (size_t)(s0 + nn) * H_DIM;
    const __bf16* kp1 = kb + (size_t)(s0 + 16 + nn) * H_DIM;
    v16bf kf00 = *(const v16bf*)(kp0 + 16*g);
    v16bf kf01 = *(const v16bf*)(kp0 + 32 + 16*g);
    v16bf kf10 = *(const v16bf*)(kp1 + 16*g);
    v16bf kf11 = *(const v16bf*)(kp1 + 32 + 16*g);
    v8f sA = vz; sA = wmma_bf16(qa0, kf00, sA); sA = wmma_bf16(qa1, kf01, sA);
    v8f sB = vz; sB = wmma_bf16(qa0, kf10, sB); sB = wmma_bf16(qa1, kf11, sB);

    // ---- S tiles 2,3 (keys s0+32..s0+63) ----
    const __bf16* kp2 = kb + (size_t)(s0 + 32 + nn) * H_DIM;
    const __bf16* kp3 = kb + (size_t)(s0 + 48 + nn) * H_DIM;
    v16bf kf20 = *(const v16bf*)(kp2 + 16*g);
    v16bf kf21 = *(const v16bf*)(kp2 + 32 + 16*g);
    v16bf kf30 = *(const v16bf*)(kp3 + 16*g);
    v16bf kf31 = *(const v16bf*)(kp3 + 32 + 16*g);
    v8f sC = vz; sC = wmma_bf16(qa0, kf20, sC); sC = wmma_bf16(qa1, kf21, sC);
    v8f sD = vz; sD = wmma_bf16(qa0, kf30, sD); sD = wmma_bf16(qa1, kf31, sD);

    // ---- mask (scale already folded into Q) ----
    float pA[8], pB[8], pC[8], pD[8];
#pragma unroll
    for (int r = 0; r < 8; ++r) {
      const unsigned lo = (unsigned)mw[r];
      const unsigned hi = (unsigned)(mw[r] >> 32);
      float a = sA[r]; if (((lo >> nn) & 1u) == 0u)        a = NEG_INF;
      float c = sB[r]; if (((lo >> (16 + nn)) & 1u) == 0u) c = NEG_INF;
      float e = sC[r]; if (((hi >> nn) & 1u) == 0u)        e = NEG_INF;
      float h = sD[r]; if (((hi >> (16 + nn)) & 1u) == 0u) h = NEG_INF;
      pA[r] = a; pB[r] = c; pC[r] = e; pD[r] = h;
    }

    asm volatile("s_wait_dscnt 0" ::: "memory");  // previous iter's LDS reads done

    // ---- online softmax update (per row, reduce over 16 lanes of half-wave) ----
#pragma unroll
    for (int r = 0; r < 8; ++r) {
      float mx = fmaxf(fmaxf(pA[r], pB[r]), fmaxf(pC[r], pD[r]));
      mx = fmaxf(mx, __shfl_xor(mx, 1, 32));
      mx = fmaxf(mx, __shfl_xor(mx, 2, 32));
      mx = fmaxf(mx, __shfl_xor(mx, 4, 32));
      mx = fmaxf(mx, __shfl_xor(mx, 8, 32));
      const float mnew  = fmaxf(mrow[r], mx);
      const float alpha = __expf(mrow[r] - mnew);
      const float ea = __expf(pA[r] - mnew);
      const float eb = __expf(pB[r] - mnew);
      const float ec = __expf(pC[r] - mnew);
      const float ed = __expf(pD[r] - mnew);
      float rs = (ea + eb) + (ec + ed);
      rs += __shfl_xor(rs, 1, 32);
      rs += __shfl_xor(rs, 2, 32);
      rs += __shfl_xor(rs, 4, 32);
      rs += __shfl_xor(rs, 8, 32);
      lrow[r] = lrow[r] * alpha + rs;
      mrow[r] = mnew;
      o[0][r] *= alpha; o[1][r] *= alpha; o[2][r] *= alpha; o[3][r] *= alpha;
      pbuf[wid][r + 8*g][nn]      = (__bf16)ea;   // P in (m, k) layout for A-frag
      pbuf[wid][r + 8*g][16 + nn] = (__bf16)eb;
      pbuf[wid][r + 8*g][32 + nn] = (__bf16)ec;
      pbuf[wid][r + 8*g][48 + nn] = (__bf16)ed;
    }

    asm volatile("s_wait_dscnt 0" ::: "memory");  // P tile visible to own wave

    // ---- P A-frags from LDS (k 0..31 and 32..63) ----
    const __bf16* pr = &pbuf[wid][nn][0];
    v16bf pa0 = cat8(*(const v8bf*)(pr + 16*g),      *(const v8bf*)(pr + 8 + 16*g));
    v16bf pa1 = cat8(*(const v8bf*)(pr + 32 + 16*g), *(const v8bf*)(pr + 40 + 16*g));

    // second-half V frags issued before first O batch so latency hides under WMMAs
    v16bf vfB[4];
#pragma unroll
    for (int nt = 0; nt < 4; ++nt)
      vfB[nt] = *(const v16bf*)(vb + (size_t)(nt*16 + nn) * T_DIM + s0 + 32 + 16*g);

    // ---- O += P x V ----
#pragma unroll
    for (int nt = 0; nt < 4; ++nt) o[nt] = wmma_bf16(pa0, vfA[nt], o[nt]);
#pragma unroll
    for (int nt = 0; nt < 4; ++nt) o[nt] = wmma_bf16(pa1, vfB[nt], o[nt]);
  }

  // ---- normalize + store f32 output [B,T,64] ----
#pragma unroll
  for (int r = 0; r < 8; ++r) {
    const float rl = __builtin_amdgcn_rcpf(lrow[r]);
#pragma unroll
    for (int nt = 0; nt < 4; ++nt) {
      out[((size_t)(b * T_DIM + t0 + r + 8*g)) * H_DIM + nt*16 + nn] = o[nt][r] * rl;
    }
  }
}

// ---------------- launch ----------------
extern "C" void kernel_launch(void* const* d_in, const int* in_sizes, int n_in,
                              void* d_out, int out_size, void* d_ws, size_t ws_size,
                              hipStream_t stream) {
  const float* x  = (const float*)d_in[0];
  const float* Wq = (const float*)d_in[1];
  const float* Wk = (const float*)d_in[2];
  const float* Wv = (const float*)d_in[3];
  const int* mask = (const int*)d_in[4];
  float* out      = (float*)d_out;

  char* ws = (char*)d_ws;
  // workspace layout (bytes):
  //   0        Wq bf16    (128 KB)
  //   131072   Wk bf16    (128 KB)
  //   262144   Wv bf16    (128 KB)
  //   393216   Q  bf16    [B*T,64]   (2 MB)
  //   2490368  K  bf16    [B*T,64]   (2 MB)
  //   4587520  Vt bf16    [B,64,T]   (2 MB)
  //   6684672  mbits u32  [T,T/32]   (2 MB)   total ~8.8 MB
  __bf16*   wqb   = (__bf16*)(ws + 0);
  __bf16*   wkb   = (__bf16*)(ws + 131072);
  __bf16*   wvb   = (__bf16*)(ws + 262144);
  __bf16*   Qb    = (__bf16*)(ws + 393216);
  __bf16*   Kb    = (__bf16*)(ws + 2490368);
  __bf16*   Vtb   = (__bf16*)(ws + 4587520);
  unsigned* mbits = (unsigned*)(ws + 6684672);

  wconv_kernel<<<(H_DIM * D_DIM + 255) / 256, 256, 0, stream>>>(Wq, Wk, Wv, wqb, wkb, wvb);
  mpack_kernel<<<(T_DIM * T_DIM / 32 + 255) / 256, 256, 0, stream>>>(mask, mbits);
  qkv_proj_kernel<<<256, 128, 0, stream>>>(x, wqb, wkb, wvb, Qb, Kb, Vtb);
  flash_attn_kernel<<<256, 128, 0, stream>>>(Qb, Kb, Vtb, mbits, out);
}
